// AttentionalLaneLSTM_30786325578420
// MI455X (gfx1250) — compile-verified, matrix operand order checked by reading, not run on hardware
//
#include <hip/hip_runtime.h>
#include <hip/hip_bf16.h>

// =====================================================================
// AttentionalLaneLSTM for MI455X (gfx1250, wave32, WMMA + TDM)
//
// 161 GFLOP of recurrent GEMM -> bf16 WMMA 16x16x32 (f32 accum).
// Each block owns 32 rows for all 100 steps of one direction; weights
// live in VGPRs (compiler-cached), hidden state in LDS, and the per-
// step x tile (32x64 bf16, contiguous) is staged one step ahead into
// an LDS double buffer by the Tensor Data Mover (tensor_load_to_lds,
// synchronized with s_wait_tensorcnt) so global latency is hidden.
// =====================================================================

typedef __attribute__((ext_vector_type(16))) __bf16 v16bf;
typedef __attribute__((ext_vector_type(8)))  float  v8f;

union Frag16 {
    v16bf    v;
    unsigned d[8];
    uint4    q[2];
};

// float -> bf16 bits, round-to-nearest-even
__device__ __forceinline__ unsigned f2bf(float f) {
    unsigned u = __float_as_uint(f);
    return (u + 0x7fffu + ((u >> 16) & 1u)) >> 16;
}

__device__ __forceinline__ float sigmoidf_(float x) {
    return 1.0f / (1.0f + __expf(-x));
}

// 16-bit A-matrix 16x32 WMMA layout: lane L holds row M=L%16.
// VGPR v (packed K pair): K = (v/4)*16 + (L/16)*8 + (v%4)*2.
__device__ __forceinline__ int klocal(int lane, int v) {
    return (v >> 2) * 16 + ((lane >> 4) & 1) * 8 + (v & 3) * 2;
}

// ---------------- Tensor Data Mover (gfx1250) ------------------------
#if defined(__has_builtin)
#if __has_builtin(__builtin_amdgcn_tensor_load_to_lds)
#define HAVE_TDM 1
#endif
#endif
#ifndef HAVE_TDM
#define HAVE_TDM 0
#endif

#if HAVE_TDM
typedef __attribute__((ext_vector_type(4))) unsigned int v4u;
typedef __attribute__((ext_vector_type(8))) int          v8i;
typedef __attribute__((ext_vector_type(4))) int          v4i;

// Issue an async TDM load of one x tile: 32 rows x 64 bf16 (row stride
// 64 elems) from global -> LDS. D# group0/group1 per CDNA5 ISA 8.3/8.4.
__device__ __forceinline__ void tdm_load_x(const unsigned short* gptr,
                                           unsigned lds_addr) {
    unsigned long long ga = (unsigned long long)gptr;
    v4u g0;
    g0[0] = 1u;                                   // count=1 (valid), user mode
    g0[1] = lds_addr;                             // lds_addr (bytes)
    g0[2] = (unsigned)(ga & 0xffffffffu);         // global_addr[31:0]
    g0[3] = (unsigned)((ga >> 32) & 0x01ffffffu)  // global_addr[56:32]
            | (2u << 30);                         // type = 2 ("image")
    v8i g1;
    g1[0] = 0x00010000;            // workgroup_mask=0, data_size=1 (2B)
    g1[1] = (64 << 16);            // tensor_dim0 = 64 (bits 79:48, low half)
    g1[2] = (int)0x40000000;       // tensor_dim0 hi=0; tensor_dim1 lo16 (409600)
    g1[3] = 6 | (64 << 16);        // tensor_dim1 hi (409600>>16); tile_dim0=64
    g1[4] = 32;                    // tile_dim1 = 32 rows; tile_dim2 = 0
    g1[5] = 64;                    // tensor_dim0_stride = 64
    g1[6] = 0;
    g1[7] = 0;
    v4i z = {0, 0, 0, 0};          // groups 2/3 disabled (2D tensor)
#if __clang_major__ >= 23
    v8i z8 = {0, 0, 0, 0, 0, 0, 0, 0};
    __builtin_amdgcn_tensor_load_to_lds(g0, g1, z, z, z8, 0);
#else
    __builtin_amdgcn_tensor_load_to_lds(g0, g1, z, z, 0);
#endif
}
#endif

// ---------------- workspace layout (bytes) ---------------------------
static const size_t X_OFF    = 0;                    // bf16[100][4096][64]
static const size_t WC_OFF   = 52428800;             // u32 [2][6][32][32][8]
static const size_t ENCW_OFF = WC_OFF + 393216;      // u32 [32][8][32][8]
static const size_t SC_OFF   = ENCW_OFF + 262144;    // f32 [256][100]
static const size_t OBF_OFF  = SC_OFF + 102400;      // bf16[4096][1024]

// =====================================================================
// Kernel 1: pre-swizzle weights into WMMA B-fragment order (bf16 pairs)
// =====================================================================
__global__ void prep_kernel(const float* __restrict__ WihF,
                            const float* __restrict__ WhhF,
                            const float* __restrict__ WihB,
                            const float* __restrict__ WhhB,
                            const float* __restrict__ encW,
                            unsigned* __restrict__ wc,
                            unsigned* __restrict__ encw) {
    const int WC_DW = 2 * 6 * 32 * 32 * 8;   // 98304
    const int EW_DW = 32 * 8 * 32 * 8;       // 65536
    int id = blockIdx.x * blockDim.x + threadIdx.x;
    if (id < WC_DW) {
        int v = id & 7;
        int t = id >> 3;
        int lane = t & 31; t >>= 5;
        int nt = t & 31;   t >>= 5;
        int kt = t % 6;
        int dd = t / 6;
        int k = kt * 32 + klocal(lane, v);   // even
        int n = nt * 16 + (lane & 15);
        const float* Wih = dd ? WihB : WihF;
        const float* Whh = dd ? WhhB : WhhF;
        float lo = (k < 64) ? Wih[k * 512 + n] : Whh[(k - 64) * 512 + n];
        float hi = (k + 1 < 64) ? Wih[(k + 1) * 512 + n] : Whh[(k + 1 - 64) * 512 + n];
        wc[id] = f2bf(lo) | (f2bf(hi) << 16);
    } else if (id < WC_DW + EW_DW) {
        int id2 = id - WC_DW;
        int v = id2 & 7;
        int t = id2 >> 3;
        int lane = t & 31; t >>= 5;
        int nt = t & 7;    t >>= 3;
        int kt = t;                          // 0..31
        int k = kt * 32 + klocal(lane, v);
        int n = nt * 16 + (lane & 15);
        float lo = encW[k * 128 + n];
        float hi = encW[(k + 1) * 128 + n];
        encw[id2] = f2bf(lo) | (f2bf(hi) << 16);
    }
}

// =====================================================================
// Kernel 2: lane embedding  x[t][m][e] = relu(feat @ embed_W + b), bf16
// =====================================================================
__global__ void embed_kernel(const float* __restrict__ lf,
                             const float* __restrict__ eW,
                             const float* __restrict__ eb,
                             unsigned short* __restrict__ x_bf) {
    int id = blockIdx.x * 256 + threadIdx.x;     // 4096*100*64 threads
    int e = id & 63;
    int mt = id >> 6;
    int t = mt % 100;
    int m = mt / 100;
    if (m >= 4096) return;
    const float* f = lf + m * 600 + 200 + t * 4;
    float acc = eb[e];
    acc += f[0] * eW[e] + f[1] * eW[64 + e] + f[2] * eW[128 + e] + f[3] * eW[192 + e];
    acc = fmaxf(acc, 0.0f);
    x_bf[(t * 4096 + m) * 64 + e] = (unsigned short)f2bf(acc);
}

// =====================================================================
// Kernel 3: attention scores = softmax(relu(obs @ attn_W + b), axis=1)
// =====================================================================
__global__ void attn_kernel(const float* __restrict__ obs,
                            const float* __restrict__ aW,
                            const float* __restrict__ ab,
                            float* __restrict__ scores) {
    __shared__ float s[100];
    __shared__ float red[128];
    int r = blockIdx.x;
    int c = threadIdx.x;
    float val = -3.0e38f;
    if (c < 100) {
        float acc = ab[c];
        for (int k = 0; k < 128; ++k) acc += obs[r * 128 + k] * aW[k * 100 + c];
        acc = fmaxf(acc, 0.0f);
        s[c] = acc;
        val = acc;
    }
    red[c] = val;
    __syncthreads();
    for (int st = 64; st > 0; st >>= 1) {
        if (c < st) red[c] = fmaxf(red[c], red[c + st]);
        __syncthreads();
    }
    float mx = red[0];
    __syncthreads();
    float ex = 0.0f;
    if (c < 100) ex = __expf(s[c] - mx);
    red[c] = ex;
    __syncthreads();
    for (int st = 64; st > 0; st >>= 1) {
        if (c < st) red[c] += red[c + st];
        __syncthreads();
    }
    float inv = 1.0f / red[0];
    if (c < 100) scores[r * 100 + c] = ex * inv;
}

// =====================================================================
// Kernel 4: bidirectional LSTM with TDM-staged x and fused reductions.
// grid = (M/32, 2 dirs), block = 256 threads (8 waves).
// =====================================================================
__global__ __launch_bounds__(256) void lstm_kernel(
    const unsigned short* __restrict__ x_bf,
    const unsigned* __restrict__ wc,
    const float* __restrict__ b_f, const float* __restrict__ b_b,
    const float* __restrict__ h0,  const float* __restrict__ c0,
    const float* __restrict__ scores, const int* __restrict__ mask,
    unsigned short* __restrict__ out_bf) {
    __shared__ float          gates[16 * 512];     // 32 KB
    __shared__ unsigned short hbuf[32 * 128];      //  8 KB bf16 hidden state
    __shared__ unsigned short xstage[2][32 * 64];  //  8 KB x double buffer

    const int tid  = threadIdx.x;
    const int lane = tid & 31;
    const int wave = tid >> 5;
    const int d    = blockIdx.y;
    const int m0   = blockIdx.x * 32;
    const float* bias = d ? b_b : b_f;

    for (int e = tid; e < 32 * 128; e += 256)
        hbuf[e] = (unsigned short)f2bf(h0[d * 128 + (e & 127)]);

    float c_reg[16], mx_reg[16], at_reg[16];
#pragma unroll
    for (int q = 0; q < 16; ++q) {
        int r = q & 7;
        int e = r * 256 + tid;
        int j = e & 127;
        c_reg[q]  = c0[d * 128 + j];
        mx_reg[q] = -3.0e38f;
        at_reg[q] = 0.0f;
    }
    __syncthreads();

    const unsigned* hb32 = (const unsigned*)hbuf;
    const uint4*    wq   = (const uint4*)(wc + d * 6 * 32 * 32 * 8);
    __builtin_prefetch((const void*)wq, 0, 1);     // global_prefetch_b8

#if HAVE_TDM
    // stage the first x tile asynchronously (wave 0 drives the TDM)
    if (wave == 0) {
        const int tt0 = d ? 99 : 0;
        tdm_load_x(x_bf + (size_t)(tt0 * 4096 + m0) * 64,
                   (unsigned)(size_t)&xstage[0][0]);
    }
#else
    const unsigned* xb32 = (const unsigned*)x_bf;
#endif

    for (int it = 0; it < 100; ++it) {
        const int tt = d ? (99 - it) : it;
#if HAVE_TDM
        __builtin_amdgcn_s_wait_tensorcnt(0);      // current tile landed
        __syncthreads();
        if (wave == 0 && it < 99) {                // prefetch next tile
            const int ttn = d ? (99 - (it + 1)) : (it + 1);
            tdm_load_x(x_bf + (size_t)(ttn * 4096 + m0) * 64,
                       (unsigned)(size_t)&xstage[(it + 1) & 1][0]);
        }
        const unsigned* xs32 = (const unsigned*)&xstage[it & 1][0];
#endif
#pragma unroll
        for (int mi = 0; mi < 2; ++mi) {
            const int nt_base = wave * 4;
            v8f acc[4];
#pragma unroll
            for (int j = 0; j < 4; ++j)
#pragma unroll
                for (int q = 0; q < 8; ++q) acc[j][q] = 0.0f;

            // preload ALL six A fragments, then one uninterrupted WMMA burst
            Frag16 afr[6];
#pragma unroll
            for (int kt = 0; kt < 2; ++kt) {       // x part
#pragma unroll
                for (int v = 0; v < 8; ++v) {
                    int k = kt * 32 + klocal(lane, v);
                    int rl = mi * 16 + (lane & 15);
#if HAVE_TDM
                    afr[kt].d[v] = xs32[rl * 32 + (k >> 1)];
#else
                    afr[kt].d[v] = xb32[(tt * 4096 + m0 + rl) * 32 + (k >> 1)];
#endif
                }
            }
#pragma unroll
            for (int kt = 2; kt < 6; ++kt) {       // h part (LDS)
#pragma unroll
                for (int v = 0; v < 8; ++v) {
                    int k = kt * 32 + klocal(lane, v);
                    afr[kt].d[v] = hb32[(mi * 16 + (lane & 15)) * 64 + ((k - 64) >> 1)];
                }
            }
#pragma unroll
            for (int kt = 0; kt < 6; ++kt) {
#pragma unroll
                for (int j = 0; j < 4; ++j) {
                    Frag16 b;
                    const uint4* bp = wq + (size_t)((kt * 32 + nt_base + j) * 32 + lane) * 2;
                    b.q[0] = bp[0];
                    b.q[1] = bp[1];
                    acc[j] = __builtin_amdgcn_wmma_f32_16x16x32_bf16(
                        false, afr[kt].v, false, b.v, (short)0, acc[j], false, false);
                }
            }
            // stage gate tile to LDS (C layout: row = v + 8*(lane/16))
#pragma unroll
            for (int j = 0; j < 4; ++j) {
                int colb = (nt_base + j) * 16 + (lane & 15);
                int rowb = 8 * (lane >> 4);
#pragma unroll
                for (int v = 0; v < 8; ++v)
                    gates[(rowb + v) * 512 + colb] = acc[j][v];
            }
            __syncthreads();

            // pointwise LSTM for subtile mi
#pragma unroll
            for (int r = 0; r < 8; ++r) {
                int e  = r * 256 + tid;
                int ml = e >> 7;
                int j  = e & 127;
                int q  = mi * 8 + r;
                float gi = gates[ml * 512 + j]       + bias[j];
                float gf = gates[ml * 512 + 128 + j] + bias[128 + j];
                float gg = gates[ml * 512 + 256 + j] + bias[256 + j];
                float go = gates[ml * 512 + 384 + j] + bias[384 + j];
                float c = sigmoidf_(gf) * c_reg[q] + sigmoidf_(gi) * tanhf(gg);
                c_reg[q] = c;
                float h = sigmoidf_(go) * tanhf(c);
                int grow = m0 + mi * 16 + ml;
                hbuf[(mi * 16 + ml) * 128 + j] = (unsigned short)f2bf(h);
                mx_reg[q] = fmaxf(mx_reg[q], h);
                float sc = scores[mask[grow] * 100 + tt];
                at_reg[q] += sc * h;
                if (tt == 0)
                    out_bf[grow * 1024 + d * 128 + j] = (unsigned short)f2bf(h);
                if (tt == 99)
                    out_bf[grow * 1024 + 256 + d * 128 + j] = (unsigned short)f2bf(h);
            }
            __syncthreads();
        }
    }
#pragma unroll
    for (int q = 0; q < 16; ++q) {
        int r = q & 7, mi = q >> 3;
        int e = r * 256 + tid;
        int ml = e >> 7;
        int j = e & 127;
        int grow = m0 + mi * 16 + ml;
        out_bf[grow * 1024 + 512 + d * 128 + j] = (unsigned short)f2bf(mx_reg[q]);
        out_bf[grow * 1024 + 768 + d * 128 + j] = (unsigned short)f2bf(at_reg[q]);
    }
}

// =====================================================================
// Kernel 5: final encoder  out = relu(out_bf(4096x1024) @ enc_W + b)
// =====================================================================
__global__ __launch_bounds__(256) void enc_kernel(
    const unsigned short* __restrict__ out_bf,
    const unsigned* __restrict__ encw,
    const float* __restrict__ eb,
    float* __restrict__ out) {
    int tid = threadIdx.x;
    int lane = tid & 31;
    int wave = tid >> 5;
    int m0 = blockIdx.x * 32;
    const unsigned* ob32 = (const unsigned*)out_bf;
    const uint4* wq = (const uint4*)encw;
    __builtin_prefetch((const void*)wq, 0, 1);

    v8f acc[2];
#pragma unroll
    for (int mi = 0; mi < 2; ++mi)
#pragma unroll
        for (int q = 0; q < 8; ++q) acc[mi][q] = 0.0f;

    for (int kt = 0; kt < 32; ++kt) {
        Frag16 b;
        const uint4* bp = wq + (size_t)((kt * 8 + wave) * 32 + lane) * 2;
        b.q[0] = bp[0];
        b.q[1] = bp[1];
#pragma unroll
        for (int mi = 0; mi < 2; ++mi) {
            Frag16 a;
#pragma unroll
            for (int v = 0; v < 8; ++v) {
                int k = kt * 32 + klocal(lane, v);
                int grow = m0 + mi * 16 + (lane & 15);
                a.d[v] = ob32[grow * 512 + (k >> 1)];
            }
            acc[mi] = __builtin_amdgcn_wmma_f32_16x16x32_bf16(
                false, a.v, false, b.v, (short)0, acc[mi], false, false);
        }
    }
    int col = wave * 16 + (lane & 15);
    float bv = eb[col];
#pragma unroll
    for (int mi = 0; mi < 2; ++mi)
#pragma unroll
        for (int v = 0; v < 8; ++v) {
            int row = m0 + mi * 16 + v + 8 * (lane >> 4);
            out[row * 128 + col] = fmaxf(acc[mi][v] + bv, 0.0f);
        }
}

// =====================================================================
extern "C" void kernel_launch(void* const* d_in, const int* in_sizes, int n_in,
                              void* d_out, int out_size, void* d_ws, size_t ws_size,
                              hipStream_t stream) {
    const float* lane_features = (const float*)d_in[0];
    const float* obs_encoding  = (const float*)d_in[1];
    const float* embed_W       = (const float*)d_in[2];
    const float* embed_b       = (const float*)d_in[3];
    const float* attn_W        = (const float*)d_in[4];
    const float* attn_b        = (const float*)d_in[5];
    const float* Wih_f         = (const float*)d_in[6];
    const float* Whh_f         = (const float*)d_in[7];
    const float* b_f           = (const float*)d_in[8];
    const float* Wih_b         = (const float*)d_in[9];
    const float* Whh_b         = (const float*)d_in[10];
    const float* b_b           = (const float*)d_in[11];
    const float* h0            = (const float*)d_in[12];
    const float* c0            = (const float*)d_in[13];
    const float* enc_W         = (const float*)d_in[14];
    const float* enc_b         = (const float*)d_in[15];
    const int*   mask          = (const int*)d_in[16];

    char* ws = (char*)d_ws;
    unsigned short* x_bf   = (unsigned short*)(ws + X_OFF);
    unsigned*       wc     = (unsigned*)(ws + WC_OFF);
    unsigned*       encw   = (unsigned*)(ws + ENCW_OFF);
    float*          scores = (float*)(ws + SC_OFF);
    unsigned short* out_bf = (unsigned short*)(ws + OBF_OFF);

    prep_kernel<<<640, 256, 0, stream>>>(Wih_f, Whh_f, Wih_b, Whh_b, enc_W, wc, encw);
    embed_kernel<<<102400, 256, 0, stream>>>(lane_features, embed_W, embed_b, x_bf);
    attn_kernel<<<256, 128, 0, stream>>>(obs_encoding, attn_W, attn_b, scores);
    lstm_kernel<<<dim3(128, 2), 256, 0, stream>>>(x_bf, wc, b_f, b_b, h0, c0,
                                                  scores, mask, out_bf);
    enc_kernel<<<128, 256, 0, stream>>>(out_bf, encw, enc_b, (float*)d_out);
}
